// STACAME_minibatch_large_77644418777396
// MI455X (gfx1250) — matrix operand order repaired and checked by reading, not compiled, not run on hardware
//
#include <hip/hip_runtime.h>
#include <hip/hip_bf16.h>
#include <math.h>

// ---------------- problem constants (match reference) ----------------
#define N_NODES   100000
#define N_EDGES   1600000
#define IN_DIM    512
#define HID       128
#define OUT_DIM   32
#define NEG_SLOPE 0.2f

typedef __attribute__((ext_vector_type(16))) __bf16 v16bf;
typedef __attribute__((ext_vector_type(8)))  float  v8f;

__device__ __forceinline__ unsigned short f2bf_bits(float x) {
  union { float f; unsigned u; } in; in.f = x;
  unsigned r = in.u + 0x7FFFu + ((in.u >> 16) & 1u);   // round-to-nearest-even
  return (unsigned short)(r >> 16);
}

// ---------------- fp32 -> bf16 row-major cast (A operand mirror) -----
__global__ void cast_f32_bf16(const float* __restrict__ s,
                              unsigned short* __restrict__ d, long long n) {
  long long i = (long long)blockIdx.x * blockDim.x + threadIdx.x;
  const long long stride = (long long)gridDim.x * blockDim.x;
  for (; i < n; i += stride) d[i] = f2bf_bits(s[i]);
}

// ---------------- pack weight matrix into WMMA B-fragment order ------
// Bp[((ks*nTiles + nt)*32 + lane)*16 + e] = B_logical[k][n]
//   n = nt*16 + (lane&15),  k = ks*32 + 8*(lane>>4) + (e&7) + 16*(e>>3)
// BT==0: B_logical[k][n] = B[k*N + n]   (K x N row-major)
// BT==1: B_logical[k][n] = B[n*K + k]   (N x K row-major, i.e. use B^T)
template <int BT>
__global__ void pack_B(const float* __restrict__ B, unsigned short* __restrict__ P,
                       int K, int N) {
  const int i = blockIdx.x * blockDim.x + threadIdx.x;
  const int total = K * N;
  if (i >= total) return;
  const int e    = i & 15;
  const int lane = (i >> 4) & 31;
  const int t    = i >> 9;           // ks*nTiles + nt
  const int nTiles = N >> 4;
  const int ks = t / nTiles, nt = t % nTiles;
  const int n = nt * 16 + (lane & 15);
  const int k = ks * 32 + 8 * (lane >> 4) + (e & 7) + 16 * (e >> 3);
  const float v = BT ? B[(long long)n * K + k] : B[(long long)k * N + n];
  P[i] = f2bf_bits(v);
}

// ---------------------------------------------------------------------
// WMMA GEMM, bf16 operands / f32 accumulate:  C[M,N] = Abf[M,K] * Bp
// Abf: row-major bf16.  Bp: fragment-packed (see pack_B).
// 4 waves/block; each wave computes a 16 x (16*NT) tile.
// Per k-step: A fragment = 2 contiguous b128 loads (ISA 7.12.2 runs at
// k=8h and k=16+8h), each B fragment = 2 contiguous b128 loads, NT wmma.
// ---------------------------------------------------------------------
template <int NT>
__global__ __launch_bounds__(128)
void gemm_wmma_packed(const unsigned short* __restrict__ Abf,
                      const unsigned short* __restrict__ Bp,
                      float* __restrict__ C, int M, int K, int N) {
  const int w    = threadIdx.x >> 5;
  const int lane = threadIdx.x & 31;
  const int nTiles = N >> 4;
  const int nCols  = nTiles / NT;
  const long long wid   = (long long)blockIdx.x * 4 + w;
  const long long total = (long long)(M >> 4) * nCols;
  if (wid >= total) return;
  const int mBase = (int)(wid / nCols) << 4;
  const int nt0   = (int)(wid % nCols) * NT;

  const int h = lane >> 4, l15 = lane & 15;

  v8f acc[NT];
#pragma unroll
  for (int t = 0; t < NT; ++t) {
    v8f z = {0.f, 0.f, 0.f, 0.f, 0.f, 0.f, 0.f, 0.f};
    acc[t] = z;
  }

  // A fragment lane base: row m = mBase + l15, runs at k=8h and k=16+8h
  const unsigned short* a0 = Abf + (long long)(mBase + l15) * K + 8 * h;
  const unsigned short* b0 = Bp + ((long long)nt0 * 32 + lane) * 16;
  const long long bStride = (long long)nTiles * 32 * 16;   // per k-step

  const int ksteps = K >> 5;
  for (int ks = 0; ks < ksteps; ++ks) {
    __builtin_prefetch(a0 + 32, 0, 0);     // global_prefetch_b8: next A block
    union { v16bf v; uint4 q[2]; } af;
    af.q[0] = *(const uint4*)(a0);         // k = ks*32 + 8h     .. +7
    af.q[1] = *(const uint4*)(a0 + 16);    // k = ks*32 + 16+8h  .. +7
    a0 += 32;
#pragma unroll
    for (int t = 0; t < NT; ++t) {
      const unsigned short* bp = b0 + (long long)t * (32 * 16);
      union { v16bf v; uint4 q[2]; } bf;
      bf.q[0] = *(const uint4*)(bp);
      bf.q[1] = *(const uint4*)(bp + 8);
      acc[t] = __builtin_amdgcn_wmma_f32_16x16x32_bf16(
          /*neg_a=*/false, af.v, /*neg_b=*/false, bf.v,
          /*c_mod=*/(short)0, acc[t], /*reuse_a=*/false, /*reuse_b=*/false);
    }
    b0 += bStride;
  }

#pragma unroll
  for (int t = 0; t < NT; ++t) {
    union { v8f v; float e[8]; } cf; cf.v = acc[t];
    const int nb = (nt0 + t) * 16;
#pragma unroll
    for (int r = 0; r < 8; ++r) {          // C/D layout: M = r + 8h, N = l15
      C[(long long)(mBase + r + 8 * h) * N + nb + l15] = cf.e[r];
    }
  }
}

// ---------------- per-node attention dot products (wave per node) ----
__global__ __launch_bounds__(128)
void att_dots(const float* __restrict__ x1, const float* __restrict__ att_s,
              const float* __restrict__ att_d, float* __restrict__ a_s,
              float* __restrict__ a_d, int n) {
  const int wave = (int)((blockIdx.x * (long long)blockDim.x + threadIdx.x) >> 5);
  const int lane = threadIdx.x & 31;
  if (wave >= n) return;
  const float* row = x1 + (long long)wave * HID;
  float ss = 0.f, dd = 0.f;
#pragma unroll
  for (int j = 0; j < HID; j += 32) {
    float v = row[j + lane];
    ss += v * att_s[j + lane];
    dd += v * att_d[j + lane];
  }
#pragma unroll
  for (int off = 16; off; off >>= 1) {
    ss += __shfl_xor(ss, off, 32);
    dd += __shfl_xor(dd, off, 32);
  }
  if (lane == 0) { a_s[wave] = ss; a_d[wave] = dd; }
}

// ---------------- utility fill --------------------------------------
__global__ void fill_f32(float* __restrict__ p, float v, long long n) {
  long long i = (long long)blockIdx.x * blockDim.x + threadIdx.x;
  const long long stride = (long long)gridDim.x * blockDim.x;
  for (; i < n; i += stride) p[i] = v;
}

// ---------------- edge pass 1: leaky-relu logit + segment max -------
__global__ void edge_logits_max(const long long* __restrict__ src,
                                const long long* __restrict__ dst,
                                const float* __restrict__ a_s,
                                const float* __restrict__ a_d,
                                float* __restrict__ ev, float* __restrict__ mx,
                                int E) {
  const int e = blockIdx.x * blockDim.x + threadIdx.x;
  if (e >= E) return;
  const long long s = src[e], d = dst[e];
  float v = a_s[s] + a_d[d];
  v = (v > 0.f) ? v : NEG_SLOPE * v;
  ev[e] = v;
  __hip_atomic_fetch_max(&mx[d], v, __ATOMIC_RELAXED, __HIP_MEMORY_SCOPE_AGENT);
}

// ---------------- edge pass 2: exp + segment sum --------------------
__global__ void edge_exp_sum(const long long* __restrict__ dst,
                             float* __restrict__ ev, const float* __restrict__ mx,
                             float* __restrict__ ssum, int E) {
  const int e = blockIdx.x * blockDim.x + threadIdx.x;
  if (e >= E) return;
  const long long d = dst[e];
  const float t = __expf(ev[e] - mx[d]);
  ev[e] = t;
  __hip_atomic_fetch_add(&ssum[d], t, __ATOMIC_RELAXED, __HIP_MEMORY_SCOPE_AGENT);
}

// ---------------- scatter: acc[dst] += x[src] * alpha (wave/edge) ----
__global__ __launch_bounds__(128)
void scatter_wave(const float* __restrict__ x, const long long* __restrict__ src,
                  const long long* __restrict__ dst, const float* __restrict__ ev,
                  const float* __restrict__ ssum, float* __restrict__ acc, int E) {
  const int wave = (int)((blockIdx.x * (long long)blockDim.x + threadIdx.x) >> 5);
  const int lane = threadIdx.x & 31;
  if (wave >= E) return;
  const long long s = src[wave], d = dst[wave];
  const float alpha = ev[wave] / (ssum[d] + 1e-16f);
  const float* xr = x + s * HID;
  float* ar = acc + d * HID;
#pragma unroll
  for (int j = 0; j < HID; j += 32) {
    __hip_atomic_fetch_add(&ar[j + lane], xr[j + lane] * alpha,
                           __ATOMIC_RELAXED, __HIP_MEMORY_SCOPE_AGENT);
  }
}

// ---------------- ELU in place --------------------------------------
__global__ void elu_inplace(float* __restrict__ p, long long n) {
  long long i = (long long)blockIdx.x * blockDim.x + threadIdx.x;
  const long long stride = (long long)gridDim.x * blockDim.x;
  for (; i < n; i += stride) {
    const float x = p[i];
    p[i] = (x > 0.f) ? x : (__expf(x) - 1.f);
  }
}

// ---------------------------------------------------------------------
extern "C" void kernel_launch(void* const* d_in, const int* in_sizes, int n_in,
                              void* d_out, int out_size, void* d_ws, size_t ws_size,
                              hipStream_t stream) {
  const float*     features = (const float*)d_in[0];
  const long long* ei       = (const long long*)d_in[1];   // int64 per reference
  const float*     W1       = (const float*)d_in[2];
  const float*     att_s    = (const float*)d_in[3];
  const float*     att_d    = (const float*)d_in[4];
  const float*     W2       = (const float*)d_in[5];
  const long long* src = ei;
  const long long* dst = ei + N_EDGES;

  float* h2 = (float*)d_out;                                  // N x OUT
  float* h4 = (float*)d_out + (long long)N_NODES * OUT_DIM;   // N x IN_DIM

  // ---- workspace layout ----
  // f32: x1/x3 | h-acc | a_s | a_d | max | sum | ev
  float* buf0 = (float*)d_ws;                         // N*HID (x1, later x3)
  float* buf1 = buf0 + (long long)N_NODES * HID;      // N*HID (h1/h3 accum)
  float* a_sv = buf1 + (long long)N_NODES * HID;      // N
  float* a_dv = a_sv + N_NODES;                       // N
  float* mx   = a_dv + N_NODES;                       // N
  float* ssum = mx + N_NODES;                         // N
  float* ev   = ssum + N_NODES;                       // E
  // bf16: shared A mirror (sized for largest A = features), weight packs
  unsigned short* Abf  = (unsigned short*)(ev + N_EDGES);
  unsigned short* W1p  = Abf + (long long)N_NODES * IN_DIM;   // 512*128
  unsigned short* W2p  = W1p + IN_DIM * HID;                  // 128*32
  unsigned short* W2Tp = W2p + HID * OUT_DIM;                 // 32*128
  unsigned short* W1Tp = W2Tp + OUT_DIM * HID;                // 128*512

  const dim3 blk(128);
  const int mt = N_NODES / 16;   // 6250 m-tiles
  auto gemmBlocks = [mt](int nTiles, int NT) {
    return dim3((unsigned)(((long long)mt * (nTiles / NT) + 3) / 4));
  };

  // ---- pack all weight fragments (tiny) ----
  pack_B<0><<<dim3((IN_DIM * HID + 255) / 256), dim3(256), 0, stream>>>(W1, W1p, IN_DIM, HID);
  pack_B<0><<<dim3((HID * OUT_DIM + 255) / 256), dim3(256), 0, stream>>>(W2, W2p, HID, OUT_DIM);
  pack_B<1><<<dim3((OUT_DIM * HID + 255) / 256), dim3(256), 0, stream>>>(W2, W2Tp, OUT_DIM, HID);
  pack_B<1><<<dim3((HID * IN_DIM + 255) / 256), dim3(256), 0, stream>>>(W1, W1Tp, HID, IN_DIM);

  // ---- conv1: x1 = X @ W1 ----
  cast_f32_bf16<<<dim3(8192), dim3(256), 0, stream>>>(features, Abf,
                                                      (long long)N_NODES * IN_DIM);
  gemm_wmma_packed<4><<<gemmBlocks(HID / 16, 4), blk, 0, stream>>>(
      Abf, W1p, buf0, N_NODES, IN_DIM, HID);

  // ---- attention coefficients + segment softmax ----
  att_dots<<<dim3((N_NODES + 3) / 4), blk, 0, stream>>>(buf0, att_s, att_d,
                                                        a_sv, a_dv, N_NODES);
  fill_f32<<<dim3(1024), dim3(256), 0, stream>>>(mx, -3.0e38f, N_NODES);
  fill_f32<<<dim3(1024), dim3(256), 0, stream>>>(ssum, 0.f, N_NODES);
  fill_f32<<<dim3(4096), dim3(256), 0, stream>>>(buf1, 0.f, (long long)N_NODES * HID);
  edge_logits_max<<<dim3((N_EDGES + 255) / 256), dim3(256), 0, stream>>>(
      src, dst, a_sv, a_dv, ev, mx, N_EDGES);
  edge_exp_sum<<<dim3((N_EDGES + 255) / 256), dim3(256), 0, stream>>>(
      dst, ev, mx, ssum, N_EDGES);

  // ---- h1 = elu(segment_sum(x1[src] * alpha)) ----
  scatter_wave<<<dim3((N_EDGES + 3) / 4), blk, 0, stream>>>(buf0, src, dst, ev,
                                                            ssum, buf1, N_EDGES);
  elu_inplace<<<dim3(4096), dim3(256), 0, stream>>>(buf1, (long long)N_NODES * HID);

  // ---- conv2: h2 = h1 @ W2 ----
  cast_f32_bf16<<<dim3(4096), dim3(256), 0, stream>>>(buf1, Abf,
                                                      (long long)N_NODES * HID);
  gemm_wmma_packed<2><<<gemmBlocks(OUT_DIM / 16, 2), blk, 0, stream>>>(
      Abf, W2p, h2, N_NODES, HID, OUT_DIM);

  // ---- conv3: x3 = h2 @ W2^T (K = 32) ----
  cast_f32_bf16<<<dim3(2048), dim3(256), 0, stream>>>(h2, Abf,
                                                      (long long)N_NODES * OUT_DIM);
  gemm_wmma_packed<4><<<gemmBlocks(HID / 16, 4), blk, 0, stream>>>(
      Abf, W2Tp, buf0, N_NODES, OUT_DIM, HID);

  // ---- h3 = elu(segment_sum(x3[src] * alpha)), same alpha ----
  fill_f32<<<dim3(4096), dim3(256), 0, stream>>>(buf1, 0.f, (long long)N_NODES * HID);
  scatter_wave<<<dim3((N_EDGES + 3) / 4), blk, 0, stream>>>(buf0, src, dst, ev,
                                                            ssum, buf1, N_EDGES);
  elu_inplace<<<dim3(4096), dim3(256), 0, stream>>>(buf1, (long long)N_NODES * HID);

  // ---- conv4: h4 = h3 @ W1^T ----
  cast_f32_bf16<<<dim3(4096), dim3(256), 0, stream>>>(buf1, Abf,
                                                      (long long)N_NODES * HID);
  gemm_wmma_packed<4><<<gemmBlocks(IN_DIM / 16, 4), blk, 0, stream>>>(
      Abf, W1Tp, h4, N_NODES, HID, IN_DIM);
}
